// PointNetPlusPlusSeg_33621003993621
// MI455X (gfx1250) — matrix operand order repaired and checked by reading, not compiled, not run on hardware
//
#include <hip/hip_runtime.h>

// ---------------------------------------------------------------------------
// PointNet++ segmentation forward on MI455X (gfx1250, wave32, WMMA bf16)
// GEMM: 32x64 output per wave (2x4 register blocking), 8 WMMAs per K-step.
// ---------------------------------------------------------------------------

typedef __attribute__((ext_vector_type(16))) __bf16 v16bf;
typedef __attribute__((ext_vector_type(8)))  float  v8f;

#define CEIL(a, b) (((a) + (b) - 1) / (b))

// problem constants
static constexpr int Bsz = 16;
static constexpr int N0  = 4096;
static constexpr int P1  = 512;
static constexpr int S1  = 32;
static constexpr int P2  = 128;
static constexpr int S2  = 64;
static constexpr int M1  = Bsz * P1 * S1;   // 262144
static constexpr int M2  = Bsz * P2 * S2;   // 131072
static constexpr int M3  = Bsz * P2;        // 2048
static constexpr int MF2 = Bsz * P1;        // 8192
static constexpr int MF1 = Bsz * N0;        // 65536
static constexpr float EPS_BN = 1e-5f;

// workspace layout (bytes, all 256-aligned)
static constexpr size_t OFF_STATS = 0;                         // 2*1024 f32
static constexpr size_t OFF_WB    = 8192;                      // bf16 weights
static constexpr size_t OFF_DIST  = OFF_WB    + 3u*1024*1024;  // FPS dist
static constexpr size_t OFF_IDX1  = OFF_DIST  + 262144;
static constexpr size_t OFF_L1XYZ = OFF_IDX1  + 32768;
static constexpr size_t OFF_GIDX1 = OFF_L1XYZ + 98304;
static constexpr size_t OFF_IDX2  = OFF_GIDX1 + 1048576;
static constexpr size_t OFF_L2XYZ = OFF_IDX2  + 8192;
static constexpr size_t OFF_GIDX2 = OFF_L2XYZ + 24576;
static constexpr size_t OFF_L1PTS = OFF_GIDX2 + 524288;        // (B,512,128) f32
static constexpr size_t OFF_L2PTS = OFF_L1PTS + 4194304;       // (B,128,256) f32
static constexpr size_t OFF_L3PTS = OFF_L2PTS + 2097152;       // (B,1024)    f32
static constexpr size_t OFF_L2F   = OFF_L3PTS + 65536;         // fp3 out
static constexpr size_t OFF_L1F   = OFF_L2F   + 2097152;       // fp2 out
static constexpr size_t OFF_ABUF  = OFF_L1F   + 4194304;       // bf16 activations
static constexpr size_t OFF_YBUF  = OFF_ABUF  + 44040192;      // f32 GEMM out

// ------------------------------- bf16 helpers ------------------------------
__device__ __forceinline__ unsigned short f2bf(float x) {
    union { float f; unsigned u; } v; v.f = x;
    unsigned r = v.u + 0x7FFFu + ((v.u >> 16) & 1u);
    return (unsigned short)(r >> 16);
}
__device__ __forceinline__ float bf2f(unsigned short h) {
    union { unsigned u; float f; } v; v.u = ((unsigned)h) << 16;
    return v.f;
}

// ------------------------- weight f32 -> bf16 (K-padded) -------------------
__global__ __launch_bounds__(256) void convert_w_kernel(
    const float* __restrict__ W, unsigned short* __restrict__ out,
    int cout, int cin, int kpad) {
    int t = blockIdx.x * 256 + threadIdx.x;
    if (t >= cout * kpad) return;
    int o = t / kpad, k = t % kpad;
    out[t] = f2bf(k < cin ? W[o * cin + k] : 0.0f);
}

// ------------------------------ FPS ----------------------------------------
__global__ __launch_bounds__(256) void fps_kernel(
    const float* __restrict__ xyz, int N, int npoint,
    int* __restrict__ outIdx, float* __restrict__ distWS) {
    int b = blockIdx.x, tid = threadIdx.x;
    xyz    += (size_t)b * N * 3;
    outIdx += (size_t)b * npoint;
    float* dist = distWS + (size_t)b * N;
    for (int i = tid; i < N; i += 256) dist[i] = 1e10f;
    __shared__ float smax[256];
    __shared__ int   simx[256];
    int far = 0;
    __syncthreads();
    for (int it = 0; it < npoint; ++it) {
        if (tid == 0) outIdx[it] = far;
        float cx = xyz[far * 3 + 0], cy = xyz[far * 3 + 1], cz = xyz[far * 3 + 2];
        float best = -1.0f; int bi = 0;
        for (int i = tid; i < N; i += 256) {
            float dx = xyz[i * 3 + 0] - cx, dy = xyz[i * 3 + 1] - cy, dz = xyz[i * 3 + 2] - cz;
            float d = dx * dx + dy * dy + dz * dz;
            float nd = fminf(dist[i], d);
            dist[i] = nd;
            if (nd > best) { best = nd; bi = i; }
        }
        smax[tid] = best; simx[tid] = bi;
        __syncthreads();
        for (int s = 128; s > 0; s >>= 1) {
            if (tid < s && smax[tid + s] > smax[tid]) {
                smax[tid] = smax[tid + s]; simx[tid] = simx[tid + s];
            }
            __syncthreads();
        }
        far = simx[0];
        __syncthreads();
    }
}

__global__ __launch_bounds__(256) void gather_xyz_kernel(
    const float* __restrict__ xyz, const int* __restrict__ idx,
    float* __restrict__ out, int S, int N) {
    int t = blockIdx.x * 256 + threadIdx.x;
    if (t >= Bsz * S) return;
    int b = t / S, k = idx[t];
    const float* p = xyz + ((size_t)b * N + k) * 3;
    out[t * 3 + 0] = p[0]; out[t * 3 + 1] = p[1]; out[t * 3 + 2] = p[2];
}

// ------------------------------ ball query ---------------------------------
__global__ __launch_bounds__(256) void ball_query_kernel(
    const float* __restrict__ xyz, const float* __restrict__ nxyz,
    int* __restrict__ out, int N, int S, int ns, float r2) {
    int t = blockIdx.x * 256 + threadIdx.x;
    if (t >= Bsz * S) return;
    int b = t / S;
    float cx = nxyz[t * 3 + 0], cy = nxyz[t * 3 + 1], cz = nxyz[t * 3 + 2];
    const float* p = xyz + (size_t)b * N * 3;
    int* o = out + (size_t)t * ns;
    int cnt = 0, first = -1;
    for (int k = 0; k < N; ++k) {
        float dx = p[k * 3 + 0] - cx, dy = p[k * 3 + 1] - cy, dz = p[k * 3 + 2] - cz;
        if (dx * dx + dy * dy + dz * dz <= r2) {
            if (first < 0) first = k;
            if (cnt < ns) o[cnt] = k;
            ++cnt;
        }
    }
    if (first < 0) first = 0;
    for (int j = (cnt < ns ? cnt : ns); j < ns; ++j) o[j] = first;
}

// --------------------------- A-matrix builders -----------------------------
__global__ __launch_bounds__(256) void group1_kernel(
    const float* __restrict__ xyz, const float* __restrict__ nxyz,
    const int* __restrict__ gidx, unsigned short* __restrict__ A) {
    int m = blockIdx.x * 256 + threadIdx.x;
    if (m >= M1) return;
    int j = m % S1, s = (m / S1) % P1, b = m / (S1 * P1);
    int k = gidx[((size_t)b * P1 + s) * S1 + j];
    const float* p = xyz + ((size_t)b * N0 + k) * 3;
    const float* c = nxyz + ((size_t)b * P1 + s) * 3;
    unsigned short* row = A + (size_t)m * 32;
    row[0] = f2bf(p[0] - c[0]); row[1] = f2bf(p[1] - c[1]); row[2] = f2bf(p[2] - c[2]);
    for (int q = 3; q < 32; ++q) row[q] = 0;
}

__global__ __launch_bounds__(256) void group2_kernel(
    const float* __restrict__ xyz, const float* __restrict__ nxyz,
    const int* __restrict__ gidx, const float* __restrict__ feat,
    unsigned short* __restrict__ A) {
    int m = blockIdx.x * 256 + threadIdx.x;
    if (m >= M2) return;
    int j = m % S2, s = (m / S2) % P2, b = m / (S2 * P2);
    int k = gidx[((size_t)b * P2 + s) * S2 + j];
    const float* p = xyz + ((size_t)b * P1 + k) * 3;
    const float* c = nxyz + ((size_t)b * P2 + s) * 3;
    const float* f = feat + ((size_t)b * P1 + k) * 128;
    unsigned short* row = A + (size_t)m * 160;
    row[0] = f2bf(p[0] - c[0]); row[1] = f2bf(p[1] - c[1]); row[2] = f2bf(p[2] - c[2]);
    for (int q = 0; q < 128; ++q) row[3 + q] = f2bf(f[q]);
    for (int q = 131; q < 160; ++q) row[q] = 0;
}

__global__ __launch_bounds__(256) void feat3_kernel(
    const float* __restrict__ xyz, const float* __restrict__ feat,
    unsigned short* __restrict__ A) {
    int m = blockIdx.x * 256 + threadIdx.x;
    if (m >= M3) return;
    const float* p = xyz + (size_t)m * 3;
    const float* f = feat + (size_t)m * 256;
    unsigned short* row = A + (size_t)m * 288;
    row[0] = f2bf(p[0]); row[1] = f2bf(p[1]); row[2] = f2bf(p[2]);
    for (int q = 0; q < 256; ++q) row[3 + q] = f2bf(f[q]);
    for (int q = 259; q < 288; ++q) row[q] = 0;
}

__global__ __launch_bounds__(256) void fp3_build_kernel(
    const float* __restrict__ pts1, const float* __restrict__ pts2,
    unsigned short* __restrict__ A) {
    int m = blockIdx.x * 256 + threadIdx.x;
    if (m >= M3) return;
    int b = m / P2;
    const float* f1 = pts1 + (size_t)m * 256;
    const float* f2 = pts2 + (size_t)b * 1024;
    unsigned short* row = A + (size_t)m * 1280;
    for (int q = 0; q < 256; ++q)  row[q]       = f2bf(f1[q]);
    for (int q = 0; q < 1024; ++q) row[256 + q] = f2bf(f2[q]);
}

__device__ __forceinline__ void knn3(const float* q, const float* pts, int S,
                                     int& i0, int& i1, int& i2,
                                     float& w0, float& w1, float& w2) {
    float d0 = 1e30f, d1 = 1e30f, d2 = 1e30f;
    i0 = i1 = i2 = 0;
    for (int k = 0; k < S; ++k) {
        float dx = pts[k * 3 + 0] - q[0], dy = pts[k * 3 + 1] - q[1], dz = pts[k * 3 + 2] - q[2];
        float d = dx * dx + dy * dy + dz * dz;
        if (d < d0)      { d2 = d1; i2 = i1; d1 = d0; i1 = i0; d0 = d; i0 = k; }
        else if (d < d1) { d2 = d1; i2 = i1; d1 = d; i1 = k; }
        else if (d < d2) { d2 = d; i2 = k; }
    }
    w0 = 1.0f / (d0 + 1e-8f); w1 = 1.0f / (d1 + 1e-8f); w2 = 1.0f / (d2 + 1e-8f);
    float ws = w0 + w1 + w2;
    w0 /= ws; w1 /= ws; w2 /= ws;
}

__global__ __launch_bounds__(256) void fp2_build_kernel(
    const float* __restrict__ xyz1, const float* __restrict__ xyz2,
    const float* __restrict__ pts1, const float* __restrict__ pts2,
    unsigned short* __restrict__ A) {
    int m = blockIdx.x * 256 + threadIdx.x;
    if (m >= MF2) return;
    int b = m / P1;
    int i0, i1, i2; float w0, w1, w2;
    knn3(xyz1 + (size_t)m * 3, xyz2 + (size_t)b * P2 * 3, P2, i0, i1, i2, w0, w1, w2);
    const float* f1 = pts1 + (size_t)m * 128;
    const float* g0 = pts2 + ((size_t)b * P2 + i0) * 256;
    const float* g1 = pts2 + ((size_t)b * P2 + i1) * 256;
    const float* g2 = pts2 + ((size_t)b * P2 + i2) * 256;
    unsigned short* row = A + (size_t)m * 384;
    for (int q = 0; q < 128; ++q) row[q] = f2bf(f1[q]);
    for (int q = 0; q < 256; ++q) row[128 + q] = f2bf(w0 * g0[q] + w1 * g1[q] + w2 * g2[q]);
}

__global__ __launch_bounds__(256) void fp1_build_kernel(
    const float* __restrict__ xyz1, const float* __restrict__ xyz2,
    const float* __restrict__ pts2, unsigned short* __restrict__ A) {
    int m = blockIdx.x * 256 + threadIdx.x;
    if (m >= MF1) return;
    int b = m / N0;
    int i0, i1, i2; float w0, w1, w2;
    knn3(xyz1 + (size_t)m * 3, xyz2 + (size_t)b * P1 * 3, P1, i0, i1, i2, w0, w1, w2);
    const float* g0 = pts2 + ((size_t)b * P1 + i0) * 128;
    const float* g1 = pts2 + ((size_t)b * P1 + i1) * 128;
    const float* g2 = pts2 + ((size_t)b * P1 + i2) * 128;
    unsigned short* row = A + (size_t)m * 128;
    for (int q = 0; q < 128; ++q) row[q] = f2bf(w0 * g0[q] + w1 * g1[q] + w2 * g2[q]);
}

// ---------------- WMMA bf16 GEMM + fused BN-stat accumulation --------------
// Y[M,N] = A[M,Kpad](bf16) * W[N,Kpad]^T(bf16) + bias.
// Each wave32 computes a 32x64 output block: 2 M-subtiles x 4 N-subtiles,
// 8 independent v_wmma_f32_16x16x32_bf16 per K-step (1.5 b128 loads / WMMA).
// Per-channel sum / sumsq accumulated atomically for BatchNorm.
union BF16Frag { uint4 q[2]; v16bf v; };

__global__ __launch_bounds__(256) void gemm_bf16_wmma_kernel(
    const unsigned short* __restrict__ A, const unsigned short* __restrict__ W,
    const float* __restrict__ bias, float* __restrict__ Y,
    float* __restrict__ sum, float* __restrict__ sq,
    int M, int K, int N) {
    const int tilesN = N >> 6;                       // 64-wide N supertiles
    const int tile = blockIdx.x * 8 + (threadIdx.x >> 5);
    if (tile >= (M >> 5) * tilesN) return;           // wave-uniform exit
    const int tn = tile % tilesN;
    const int tm = tile / tilesN;
    const int lane = threadIdx.x & 31;
    const int l15 = lane & 15;
    const int half = lane >> 4;

    // A fragment rows (ISA 7.12.2 16x32 bf16 layout): lanes 0-15 row = lane,
    // VGPR0-3 = K 0..7 (upper half-lanes: 8..15), VGPR4-7 = K 16..23 (24..31).
    const unsigned short* Arow0 = A + (size_t)(tm * 32 + l15) * K;
    const unsigned short* Arow1 = Arow0 + (size_t)16 * K;
    // B fragment: lane col = lane&15, same K order as A: VGPR0-3 = K 0..7
    // (upper half-lanes: 8..15), VGPR4-7 = K 16..23 (24..31).
    const unsigned short* Wrow = W + (size_t)(tn * 64 + l15) * K;

    v8f acc[2][4];
#pragma unroll
    for (int ni = 0; ni < 4; ++ni) {
        float bval = bias[tn * 64 + ni * 16 + l15];
#pragma unroll
        for (int v = 0; v < 8; ++v) { acc[0][ni][v] = bval; acc[1][ni][v] = bval; }
    }

    for (int k0 = 0; k0 < K; k0 += 32) {
        if (k0 + 32 < K) {
            __builtin_prefetch((const void*)(Arow0 + k0 + 32), 0, 1);
            __builtin_prefetch((const void*)(Arow1 + k0 + 32), 0, 1);
        }
        BF16Frag a0, a1;
        a0.q[0] = *(const uint4*)(Arow0 + k0 + half * 8);
        a0.q[1] = *(const uint4*)(Arow0 + k0 + 16 + half * 8);
        a1.q[0] = *(const uint4*)(Arow1 + k0 + half * 8);
        a1.q[1] = *(const uint4*)(Arow1 + k0 + 16 + half * 8);
#pragma unroll
        for (int ni = 0; ni < 4; ++ni) {
            BF16Frag b;
            const unsigned short* wp = Wrow + (size_t)ni * 16 * K + k0 + half * 8;
            b.q[0] = *(const uint4*)(wp);
            b.q[1] = *(const uint4*)(wp + 16);
            acc[0][ni] = __builtin_amdgcn_wmma_f32_16x16x32_bf16(
                false, a0.v, false, b.v, (short)0, acc[0][ni], false, false);
            acc[1][ni] = __builtin_amdgcn_wmma_f32_16x16x32_bf16(
                false, a1.v, false, b.v, (short)0, acc[1][ni], false, false);
        }
    }

#pragma unroll
    for (int ni = 0; ni < 4; ++ni) {
        int col = tn * 64 + ni * 16 + l15;
        float s = 0.0f, s2 = 0.0f;
#pragma unroll
        for (int mi = 0; mi < 2; ++mi) {
#pragma unroll
            for (int v = 0; v < 8; ++v) {
                // C/D layout: VGPR v -> M = v (lanes 0-15) or v+8 (lanes 16-31)
                int row = tm * 32 + mi * 16 + v + half * 8;
                float x = acc[mi][ni][v];
                Y[(size_t)row * N + col] = x;
                s += x; s2 += x * x;
            }
        }
        atomicAdd(&sum[col], s);
        atomicAdd(&sq[col],  s2);
    }
}

// ----------------------- BatchNorm apply variants --------------------------
__global__ __launch_bounds__(256) void bn_relu_bf16_kernel(
    const float* __restrict__ Y, const float* __restrict__ sum, const float* __restrict__ sq,
    const float* __restrict__ g, const float* __restrict__ be,
    unsigned short* __restrict__ A, int M, int N) {
    int t = blockIdx.x * 256 + threadIdx.x;
    if (t >= M * N) return;
    int c = t % N;
    float mean = sum[c] / (float)M;
    float var  = sq[c] / (float)M - mean * mean;
    float r = rsqrtf(var + EPS_BN);
    float v = g[c] * ((Y[t] - mean) * r) + be[c];
    A[t] = f2bf(v > 0.0f ? v : 0.0f);
}

__global__ __launch_bounds__(256) void bn_relu_f32_kernel(
    const float* __restrict__ Y, const float* __restrict__ sum, const float* __restrict__ sq,
    const float* __restrict__ g, const float* __restrict__ be,
    float* __restrict__ out, int M, int N) {
    int t = blockIdx.x * 256 + threadIdx.x;
    if (t >= M * N) return;
    int c = t % N;
    float mean = sum[c] / (float)M;
    float var  = sq[c] / (float)M - mean * mean;
    float r = rsqrtf(var + EPS_BN);
    float v = g[c] * ((Y[t] - mean) * r) + be[c];
    out[t] = v > 0.0f ? v : 0.0f;
}

__global__ __launch_bounds__(256) void bn_relu_pool_kernel(
    const float* __restrict__ Y, const float* __restrict__ sum, const float* __restrict__ sq,
    const float* __restrict__ g, const float* __restrict__ be,
    float* __restrict__ out, int groups, int G, int N) {
    int t = blockIdx.x * 256 + threadIdx.x;
    if (t >= groups * N) return;
    int c = t % N, grp = t / N;
    int M = groups * G;
    float mean = sum[c] / (float)M;
    float var  = sq[c] / (float)M - mean * mean;
    float r = rsqrtf(var + EPS_BN);
    float ga = g[c], bb = be[c];
    float best = 0.0f;   // ReLU outputs are >= 0
    for (int j = 0; j < G; ++j) {
        float v = ga * ((Y[((size_t)grp * G + j) * N + c] - mean) * r) + bb;
        v = v > 0.0f ? v : 0.0f;
        if (v > best) best = v;
    }
    out[(size_t)grp * N + c] = best;
}

// ----------------------------- final head ----------------------------------
__global__ __launch_bounds__(256) void head2_kernel(
    const unsigned short* __restrict__ A, const float* __restrict__ W,
    const float* __restrict__ bias, float* __restrict__ out) {
    int t = blockIdx.x * 256 + threadIdx.x;
    if (t >= MF1 * 13) return;
    int m = t / 13, o = t % 13;
    int b = m / N0, n = m % N0;
    const unsigned short* x = A + (size_t)m * 128;
    const float* w = W + o * 128;
    float s = bias[o];
    for (int c = 0; c < 128; ++c) s += bf2f(x[c]) * w[c];
    out[((size_t)b * 13 + o) * N0 + n] = s;
}

// ---------------------------------------------------------------------------
struct LayerDef { int cout, cin, kpad; int iW, ib, ig, ibe; };
static const LayerDef LAYERS[17] = {
    {  64,    3,   32,  1,  2,  3,  4},   // sa1.0
    {  64,   64,   64,  5,  6,  7,  8},   // sa1.1
    { 128,   64,   64,  9, 10, 11, 12},   // sa1.2
    { 128,  131,  160, 13, 14, 15, 16},   // sa2.0
    { 128,  128,  128, 17, 18, 19, 20},   // sa2.1
    { 256,  128,  128, 21, 22, 23, 24},   // sa2.2
    { 256,  259,  288, 25, 26, 27, 28},   // sa3.0
    { 512,  256,  256, 29, 30, 31, 32},   // sa3.1
    {1024,  512,  512, 33, 34, 35, 36},   // sa3.2
    { 256, 1280, 1280, 37, 38, 39, 40},   // fp3.0
    { 256,  256,  256, 41, 42, 43, 44},   // fp3.1
    { 256,  384,  384, 45, 46, 47, 48},   // fp2.0
    { 128,  256,  256, 49, 50, 51, 52},   // fp2.1
    { 128,  128,  128, 53, 54, 55, 56},   // fp1.0
    { 128,  128,  128, 57, 58, 59, 60},   // fp1.1
    { 128,  128,  128, 61, 62, 63, 64},   // fp1.2
    { 128,  128,  128, 65, 66, 67, 68},   // head1
};

extern "C" void kernel_launch(void* const* d_in, const int* in_sizes, int n_in,
                              void* d_out, int out_size, void* d_ws, size_t ws_size,
                              hipStream_t stream) {
    (void)in_sizes; (void)n_in; (void)ws_size;
    char* ws = (char*)d_ws;
    const float* xyz = (const float*)d_in[0];

    float* stats_sum = (float*)(ws + OFF_STATS);
    float* stats_sq  = stats_sum + 1024;
    unsigned short* wb   = (unsigned short*)(ws + OFF_WB);
    float* dist  = (float*)(ws + OFF_DIST);
    int*   idx1  = (int*)(ws + OFF_IDX1);
    float* l1xyz = (float*)(ws + OFF_L1XYZ);
    int*   gidx1 = (int*)(ws + OFF_GIDX1);
    int*   idx2  = (int*)(ws + OFF_IDX2);
    float* l2xyz = (float*)(ws + OFF_L2XYZ);
    int*   gidx2 = (int*)(ws + OFF_GIDX2);
    float* l1pts = (float*)(ws + OFF_L1PTS);
    float* l2pts = (float*)(ws + OFF_L2PTS);
    float* l3pts = (float*)(ws + OFF_L3PTS);
    float* l2f   = (float*)(ws + OFF_L2F);
    float* l1f   = (float*)(ws + OFF_L1F);
    unsigned short* Abuf = (unsigned short*)(ws + OFF_ABUF);
    float* Ybuf  = (float*)(ws + OFF_YBUF);

    // --- weight conversion (f32 -> bf16, K zero-padded) ---
    size_t wbOff[17]; size_t acc = 0;
    for (int i = 0; i < 17; ++i) { wbOff[i] = acc; acc += (size_t)LAYERS[i].cout * LAYERS[i].kpad; }
    for (int i = 0; i < 17; ++i) {
        const LayerDef& L = LAYERS[i];
        int total = L.cout * L.kpad;
        convert_w_kernel<<<CEIL(total, 256), 256, 0, stream>>>(
            (const float*)d_in[L.iW], wb + wbOff[i], L.cout, L.cin, L.kpad);
    }

    auto run_gemm = [&](int li, const unsigned short* Ain, int M) {
        const LayerDef& L = LAYERS[li];
        hipMemsetAsync(ws + OFF_STATS, 0, 8192, stream);
        int waves = (M / 32) * (L.cout / 64);      // 32x64 block per wave
        gemm_bf16_wmma_kernel<<<CEIL(waves, 8), 256, 0, stream>>>(
            Ain, wb + wbOff[li], (const float*)d_in[L.ib], Ybuf,
            stats_sum, stats_sq, M, L.kpad, L.cout);
    };
    auto bn_bf16 = [&](int li, int M) {
        const LayerDef& L = LAYERS[li];
        bn_relu_bf16_kernel<<<CEIL(M * L.cout, 256), 256, 0, stream>>>(
            Ybuf, stats_sum, stats_sq, (const float*)d_in[L.ig], (const float*)d_in[L.ibe],
            Abuf, M, L.cout);
    };
    auto bn_f32 = [&](int li, int M, float* out) {
        const LayerDef& L = LAYERS[li];
        bn_relu_f32_kernel<<<CEIL(M * L.cout, 256), 256, 0, stream>>>(
            Ybuf, stats_sum, stats_sq, (const float*)d_in[L.ig], (const float*)d_in[L.ibe],
            out, M, L.cout);
    };
    auto bn_pool = [&](int li, int groups, int G, float* out) {
        const LayerDef& L = LAYERS[li];
        bn_relu_pool_kernel<<<CEIL(groups * L.cout, 256), 256, 0, stream>>>(
            Ybuf, stats_sum, stats_sq, (const float*)d_in[L.ig], (const float*)d_in[L.ibe],
            out, groups, G, L.cout);
    };

    // ------------------- SA1 -------------------
    fps_kernel<<<Bsz, 256, 0, stream>>>(xyz, N0, P1, idx1, dist);
    gather_xyz_kernel<<<CEIL(Bsz * P1, 256), 256, 0, stream>>>(xyz, idx1, l1xyz, P1, N0);
    ball_query_kernel<<<CEIL(Bsz * P1, 256), 256, 0, stream>>>(xyz, l1xyz, gidx1, N0, P1, S1, 0.04f);
    group1_kernel<<<CEIL(M1, 256), 256, 0, stream>>>(xyz, l1xyz, gidx1, Abuf);
    run_gemm(0, Abuf, M1); bn_bf16(0, M1);
    run_gemm(1, Abuf, M1); bn_bf16(1, M1);
    run_gemm(2, Abuf, M1); bn_pool(2, Bsz * P1, S1, l1pts);

    // ------------------- SA2 -------------------
    fps_kernel<<<Bsz, 256, 0, stream>>>(l1xyz, P1, P2, idx2, dist);
    gather_xyz_kernel<<<CEIL(Bsz * P2, 256), 256, 0, stream>>>(l1xyz, idx2, l2xyz, P2, P1);
    ball_query_kernel<<<CEIL(Bsz * P2, 256), 256, 0, stream>>>(l1xyz, l2xyz, gidx2, P1, P2, S2, 0.16f);
    group2_kernel<<<CEIL(M2, 256), 256, 0, stream>>>(l1xyz, l2xyz, gidx2, l1pts, Abuf);
    run_gemm(3, Abuf, M2); bn_bf16(3, M2);
    run_gemm(4, Abuf, M2); bn_bf16(4, M2);
    run_gemm(5, Abuf, M2); bn_pool(5, Bsz * P2, S2, l2pts);

    // ------------------- SA3 (group-all) -------------------
    feat3_kernel<<<CEIL(M3, 256), 256, 0, stream>>>(l2xyz, l2pts, Abuf);
    run_gemm(6, Abuf, M3); bn_bf16(6, M3);
    run_gemm(7, Abuf, M3); bn_bf16(7, M3);
    run_gemm(8, Abuf, M3); bn_pool(8, Bsz, P2, l3pts);

    // ------------------- FP3 (S==1 broadcast) -------------------
    fp3_build_kernel<<<CEIL(M3, 256), 256, 0, stream>>>(l2pts, l3pts, Abuf);
    run_gemm(9, Abuf, M3);  bn_bf16(9, M3);
    run_gemm(10, Abuf, M3); bn_f32(10, M3, l2f);

    // ------------------- FP2 -------------------
    fp2_build_kernel<<<CEIL(MF2, 256), 256, 0, stream>>>(l1xyz, l2xyz, l1pts, l2f, Abuf);
    run_gemm(11, Abuf, MF2); bn_bf16(11, MF2);
    run_gemm(12, Abuf, MF2); bn_f32(12, MF2, l1f);

    // ------------------- FP1 -------------------
    fp1_build_kernel<<<CEIL(MF1, 256), 256, 0, stream>>>(xyz, l1xyz, l1f, Abuf);
    run_gemm(13, Abuf, MF1); bn_bf16(13, MF1);
    run_gemm(14, Abuf, MF1); bn_bf16(14, MF1);
    run_gemm(15, Abuf, MF1); bn_bf16(15, MF1);

    // ------------------- head -------------------
    run_gemm(16, Abuf, MF1); bn_bf16(16, MF1);
    head2_kernel<<<CEIL(MF1 * 13, 256), 256, 0, stream>>>(
        Abuf, (const float*)d_in[69], (const float*)d_in[70], (float*)d_out);
    (void)out_size;
}
